// MLP_52905407152482
// MI455X (gfx1250) — 8-bit matrix operand order repaired and checked by reading, compiled, not run on hardware
//
#include <hip/hip_runtime.h>
#include <hip/hip_bf16.h>
#include <cstdint>
#include <cstddef>


typedef int v8i __attribute__((ext_vector_type(8)));

static constexpr int   M_TOK = 2 * 4096;
static constexpr int   C_DIM = 2048;
static constexpr int   F_DIM = 8192;
static constexpr float EPS   = 1e-5f;


__device__ __forceinline__ void async_g2l_b128(void* lds_dst, const void* gsrc) {
  const uint32_t l = (uint32_t)(uintptr_t)lds_dst;
  asm volatile("global_load_async_to_lds_b128 %0, %1, off"
               :: "v"(l), "v"(gsrc)
               : "memory");
}

__device__ __forceinline__ void wait_asynccnt0() {
  asm volatile("s_wait_asynccnt 0x0" ::: "memory");
}


__device__ __forceinline__ v8i load_a_frag_lds(const int8_t* sh, int lane) {
  const int lm = lane & 15;
  const int lh = lane >> 4;
  const int8_t* p = sh + lm * 64 + lh * 16;
  v8i a;
#pragma unroll
  for (int j = 0; j < 2; ++j) {
    int4 t = *reinterpret_cast<const int4*>(p + 32 * j);
    a[4 * j]     = t.x;
    a[4 * j + 1] = t.y;
    a[4 * j + 2] = t.z;
    a[4 * j + 3] = t.w;
  }
  return a;
}

__device__ __forceinline__ v8i load_b_frag(const int8_t* __restrict__ W, int ldw,
                                           int n0, int kb, int lane) {
  const int lm = lane & 15;
  const int lh = lane >> 4;
  const int8_t* p = W + (size_t)(n0 + lm) * ldw + kb + lh * 16;
  int4 t0 = *reinterpret_cast<const int4*>(p);
  int4 t1 = *reinterpret_cast<const int4*>(p + 32);
  v8i b;
  b[0] = t0.x; b[1] = t0.y; b[2] = t0.z; b[3] = t0.w;
  b[4] = t1.x; b[5] = t1.y; b[6] = t1.z; b[7] = t1.w;
  return b;
}

__device__ __forceinline__ void stage_a_tile(int8_t* shbuf,
                                             const int8_t* __restrict__ A,
                                             int lda, int m0, int kb,
                                             int lane, int wave) {
  if (wave == 0) {
    const int lm = lane & 15;
    const int lh = lane >> 4;
    const int8_t* g = A + (size_t)(m0 + lm) * lda + kb + lh * 16;
    int8_t* d = shbuf + lm * 64 + lh * 16;
    async_g2l_b128(d, g);
    async_g2l_b128(d + 32, g + 32);
  }
}


__global__ __launch_bounds__(256) void absmean_partial_kernel(
    const float* __restrict__ w, size_t n, float* __restrict__ partial) {
  __shared__ float red[256];
  float s = 0.f;
  for (size_t i = (size_t)blockIdx.x * 256 + threadIdx.x; i < n;
       i += (size_t)gridDim.x * 256)
    s += fabsf(w[i]);
  red[threadIdx.x] = s;
  __syncthreads();
#pragma unroll
  for (int t = 128; t > 0; t >>= 1) {
    if ((int)threadIdx.x < t) red[threadIdx.x] += red[threadIdx.x + t];
    __syncthreads();
  }
  if (threadIdx.x == 0) partial[blockIdx.x] = red[0];
}

__global__ __launch_bounds__(256) void finalize_scale_kernel(
    const float* __restrict__ partial, int nparts, float inv_n,
    float* __restrict__ scale) {
  __shared__ float red[256];
  float s = 0.f;
  for (int i = threadIdx.x; i < nparts; i += 256) s += partial[i];
  red[threadIdx.x] = s;
  __syncthreads();
#pragma unroll
  for (int t = 128; t > 0; t >>= 1) {
    if ((int)threadIdx.x < t) red[threadIdx.x] += red[threadIdx.x + t];
    __syncthreads();
  }
  if (threadIdx.x == 0) scale[0] = fmaxf(red[0] * inv_n, EPS);
}


__global__ __launch_bounds__(256) void quant_weight_kernel(
    const float* __restrict__ w, const float* __restrict__ scale_p,
    int8_t* __restrict__ wq, size_t n) {
  const float inv = 1.0f / scale_p[0];
  for (size_t i = (size_t)blockIdx.x * 256 + threadIdx.x; i < n;
       i += (size_t)gridDim.x * 256) {
    float v = rintf(w[i] * inv);
    v = fminf(fmaxf(v, -1.f), 1.f);
    wq[i] = (int8_t)v;
  }
}


__global__ __launch_bounds__(256) void quant_rows_kernel(
    const float* __restrict__ x, int8_t* __restrict__ xq,
    float* __restrict__ inv_scale, int cols) {
  __shared__ float red[256];
  const int row = blockIdx.x;
  const float* xr = x + (size_t)row * cols;
  float amax = 0.f;
  for (int c = threadIdx.x; c < cols; c += 256) amax = fmaxf(amax, fabsf(xr[c]));
  red[threadIdx.x] = amax;
  __syncthreads();
#pragma unroll
  for (int t = 128; t > 0; t >>= 1) {
    if ((int)threadIdx.x < t)
      red[threadIdx.x] = fmaxf(red[threadIdx.x], red[threadIdx.x + t]);
    __syncthreads();
  }
  const float m = fmaxf(red[0], EPS);
  const float sc = 127.0f / m;
  if (threadIdx.x == 0) inv_scale[row] = m / 127.0f;
  int8_t* qr = xq + (size_t)row * cols;
  for (int c = threadIdx.x; c < cols; c += 256) {
    float v = rintf(xr[c] * sc);
    v = fminf(fmaxf(v, -128.f), 127.f);
    qr[c] = (int8_t)v;
  }
}


__global__ __launch_bounds__(128) void gemm1_swiglu_kernel(
    const int8_t* __restrict__ xq, const float* __restrict__ xs,
    const int8_t* __restrict__ wq, const float* __restrict__ wscale_p,
    float* __restrict__ h2) {
  __shared__ __align__(16) int8_t shA[2][16 * 64];

  const int lane = threadIdx.x & 31;
  const int wave = threadIdx.x >> 5;
  const int m0 = blockIdx.y * 16;
  const int n0 = blockIdx.x * 256 + wave * 64;
  const float wscale = wscale_p[0];

  const v8i zero = {0, 0, 0, 0, 0, 0, 0, 0};
  v8i accg[4] = {zero, zero, zero, zero};
  v8i accu[4] = {zero, zero, zero, zero};

  constexpr int KSTEPS = C_DIM / 64;

  stage_a_tile(shA[0], xq, C_DIM, m0, 0, lane, wave);
  if (wave == 0) wait_asynccnt0();
  __syncthreads();

  for (int i = 0; i < KSTEPS; ++i) {
    const int kb = i * 64;
    const int buf = i & 1;
    if (i + 1 < KSTEPS)
      stage_a_tile(shA[buf ^ 1], xq, C_DIM, m0, kb + 64, lane, wave);

    v8i a = load_a_frag_lds(shA[buf], lane);

    v8i bfr[8];
#pragma unroll
    for (int j = 0; j < 4; ++j) {
      bfr[j]     = load_b_frag(wq, C_DIM, n0 + 16 * j, kb, lane);
      bfr[4 + j] = load_b_frag(wq, C_DIM, F_DIM + n0 + 16 * j, kb, lane);
    }
#pragma unroll
    for (int j = 0; j < 4; ++j) {
      accg[j] = __builtin_amdgcn_wmma_i32_16x16x64_iu8(
          true, a, true, bfr[j], accg[j], false, false);
      accu[j] = __builtin_amdgcn_wmma_i32_16x16x64_iu8(
          true, a, true, bfr[4 + j], accu[j], false, false);
    }

    if (wave == 0 && i + 1 < KSTEPS) wait_asynccnt0();
    __syncthreads();
  }

  const int lm = lane & 15;
  const int lh = lane >> 4;
#pragma unroll
  for (int r = 0; r < 8; ++r) {
    const int row = m0 + r + 8 * lh;
    const float sa = xs[row] * wscale;
#pragma unroll
    for (int j = 0; j < 4; ++j) {
      const float g = (float)accg[j][r] * sa;
      const float u = (float)accu[j][r] * sa;
      const float s = g / (1.0f + __expf(-g));
      h2[(size_t)row * F_DIM + n0 + 16 * j + lm] = s * u;
    }
  }
}


__global__ __launch_bounds__(128) void gemm2_kernel(
    const int8_t* __restrict__ hq, const float* __restrict__ hs,
    const int8_t* __restrict__ wdq, const float* __restrict__ wdscale_p,
    float* __restrict__ out) {
  __shared__ __align__(16) int8_t shA[2][16 * 64];

  const int lane = threadIdx.x & 31;
  const int wave = threadIdx.x >> 5;
  const int m0 = blockIdx.y * 16;
  const int n0 = blockIdx.x * 256 + wave * 64;
  const float wdscale = wdscale_p[0];

  const v8i zero = {0, 0, 0, 0, 0, 0, 0, 0};
  v8i acc[4] = {zero, zero, zero, zero};

  constexpr int KSTEPS = F_DIM / 64;

  stage_a_tile(shA[0], hq, F_DIM, m0, 0, lane, wave);
  if (wave == 0) wait_asynccnt0();
  __syncthreads();

  for (int i = 0; i < KSTEPS; ++i) {
    const int kb = i * 64;
    const int buf = i & 1;
    if (i + 1 < KSTEPS)
      stage_a_tile(shA[buf ^ 1], hq, F_DIM, m0, kb + 64, lane, wave);

    v8i a = load_a_frag_lds(shA[buf], lane);

    v8i bfr[4];
#pragma unroll
    for (int j = 0; j < 4; ++j)
      bfr[j] = load_b_frag(wdq, F_DIM, n0 + 16 * j, kb, lane);
#pragma unroll
    for (int j = 0; j < 4; ++j)
      acc[j] = __builtin_amdgcn_wmma_i32_16x16x64_iu8(
          true, a, true, bfr[j], acc[j], false, false);

    if (wave == 0 && i + 1 < KSTEPS) wait_asynccnt0();
    __syncthreads();
  }

  const int lm = lane & 15;
  const int lh = lane >> 4;
#pragma unroll
  for (int r = 0; r < 8; ++r) {
    const int row = m0 + r + 8 * lh;
    const float sa = hs[row] * wdscale;
#pragma unroll
    for (int j = 0; j < 4; ++j) {
      out[(size_t)row * C_DIM + n0 + 16 * j + lm] = (float)acc[j][r] * sa;
    }
  }
}


extern "C" void kernel_launch(void* const* d_in, const int* in_sizes, int n_in,
                              void* d_out, int out_size, void* d_ws, size_t ws_size,
                              hipStream_t stream) {
  (void)in_sizes; (void)n_in; (void)out_size; (void)ws_size;

  const float* x    = (const float*)d_in[0];
  const float* w_gu = (const float*)d_in[1];
  const float* w_dn = (const float*)d_in[2];
  float* out = (float*)d_out;

  uintptr_t base = (uintptr_t)d_ws;
  size_t cur = 0;
  auto take = [&](size_t bytes) -> void* {
    cur = (cur + 255) & ~(size_t)255;
    void* p = (void*)(base + cur);
    cur += bytes;
    return p;
  };

  const size_t n_wgu = (size_t)2 * F_DIM * C_DIM;
  const size_t n_wdn = (size_t)C_DIM * F_DIM;

  int8_t* wgu_q   = (int8_t*)take(n_wgu);
  int8_t* wdn_q   = (int8_t*)take(n_wdn);
  int8_t* xq      = (int8_t*)take((size_t)M_TOK * C_DIM);
  float*  xs      = (float*)take((size_t)M_TOK * 4);
  float*  h2      = (float*)take((size_t)M_TOK * F_DIM * 4);
  int8_t* h2q     = (int8_t*)take((size_t)M_TOK * F_DIM);
  float*  hs      = (float*)take((size_t)M_TOK * 4);
  float*  partial = (float*)take(1024 * 4);
  float*  sc_gu   = (float*)take(4);
  float*  sc_dn   = (float*)take(4);

  absmean_partial_kernel<<<1024, 256, 0, stream>>>(w_gu, n_wgu, partial);
  finalize_scale_kernel<<<1, 256, 0, stream>>>(partial, 1024, 1.0f / (float)n_wgu, sc_gu);
  quant_weight_kernel<<<8192, 256, 0, stream>>>(w_gu, sc_gu, wgu_q, n_wgu);

  absmean_partial_kernel<<<1024, 256, 0, stream>>>(w_dn, n_wdn, partial);
  finalize_scale_kernel<<<1, 256, 0, stream>>>(partial, 1024, 1.0f / (float)n_wdn, sc_dn);
  quant_weight_kernel<<<8192, 256, 0, stream>>>(w_dn, sc_dn, wdn_q, n_wdn);

  quant_rows_kernel<<<M_TOK, 256, 0, stream>>>(x, xq, xs, C_DIM);

  gemm1_swiglu_kernel<<<dim3(F_DIM / 256, M_TOK / 16), 128, 0, stream>>>(
      xq, xs, wgu_q, sc_gu, h2);

  quant_rows_kernel<<<M_TOK, 256, 0, stream>>>(h2, h2q, hs, F_DIM);

  gemm2_kernel<<<dim3(C_DIM / 256, M_TOK / 16), 128, 0, stream>>>(
      h2q, hs, wdn_q, sc_dn, out);
}
